// LSTMSiamesePolyPredict_54417235640482
// MI455X (gfx1250) — matrix operand order repaired and checked by reading, not compiled, not run on hardware
//
#include <hip/hip_runtime.h>
#include <math.h>

typedef __bf16 bf16_t;
typedef __attribute__((ext_vector_type(8)))  __bf16 v8bf;
typedef __attribute__((ext_vector_type(16))) __bf16 v16bf;
typedef __attribute__((ext_vector_type(8)))  float  v8f;

#define NODE_DIM_ 58

static inline int cdiv_i(long long a, long long b) { return (int)((a + b - 1) / b); }

__device__ __forceinline__ bf16_t f2bf(float f) {
  unsigned u = __float_as_uint(f);
  unsigned r = u + 0x7FFFu + ((u >> 16) & 1u);   // round-to-nearest-even
  unsigned short h = (unsigned short)(r >> 16);
  bf16_t o;
  __builtin_memcpy(&o, &h, 2);
  return o;
}

__device__ __forceinline__ float sigmoidf_(float x) { return 1.f / (1.f + __expf(-x)); }

__device__ __forceinline__ v16bf load16(const bf16_t* lo, const bf16_t* hi) {
  v8bf a = *(const v8bf*)lo;
  v8bf b = *(const v8bf*)hi;
  return __builtin_shufflevector(a, b, 0,1,2,3,4,5,6,7,8,9,10,11,12,13,14,15);
}

// ---------------------------------------------------------------------------
// Generic fill
// ---------------------------------------------------------------------------
__global__ void fill_u32_k(unsigned* p, unsigned v, long long n) {
  long long i = (long long)blockIdx.x * blockDim.x + threadIdx.x;
  long long stride = (long long)gridDim.x * blockDim.x;
  for (; i < n; i += stride) p[i] = v;
}

// ---------------------------------------------------------------------------
// f32 -> bf16 conversions
// ---------------------------------------------------------------------------
__global__ void cvt_pad_bf16_k(const float* __restrict__ src, bf16_t* __restrict__ dst,
                               int M, int Ks, int Kd) {
  long long i = (long long)blockIdx.x * blockDim.x + threadIdx.x;
  long long total = (long long)M * Kd;
  if (i >= total) return;
  int r = (int)(i / Kd), k = (int)(i % Kd);
  dst[i] = (k < Ks) ? f2bf(src[(size_t)r * Ks + k]) : f2bf(0.f);
}

// W [K, Nc] f32  ->  WT [Nc, Kp] bf16 (transposed, zero padded K)
__global__ void cvt_wt_k(const float* __restrict__ W, bf16_t* __restrict__ WT,
                         int K, int Nc, int Kp) {
  long long i = (long long)blockIdx.x * blockDim.x + threadIdx.x;
  long long total = (long long)Nc * Kp;
  if (i >= total) return;
  int n = (int)(i / Kp), k = (int)(i % Kp);
  WT[i] = (k < K) ? f2bf(W[(size_t)k * Nc + n]) : f2bf(0.f);
}

// ---------------------------------------------------------------------------
// WMMA bf16 GEMM: C[M,Ncols] = A[M,K] * BT[Ncols,K]^T   (f32 accumulate)
// One wave -> 16 rows x 64 cols. Requires M%16==0, K%32==0, Ncols%64==0.
// ---------------------------------------------------------------------------
__global__ void __launch_bounds__(256) wmma_gemm_bf16_k(
    const bf16_t* __restrict__ A, const bf16_t* __restrict__ BT,
    float* __restrict__ C, int M, int K, int Ncols) {
  int gw   = (int)(((long long)blockIdx.x * blockDim.x + threadIdx.x) >> 5);
  int lane = threadIdx.x & 31;
  int colTiles = Ncols >> 6;
  int mt = gw / colTiles;
  int ct = gw - mt * colTiles;
  if (mt * 16 >= M) return;
  int half = lane >> 4;        // 0 or 1
  int l15  = lane & 15;
  const bf16_t* arow  = A  + (size_t)(mt * 16 + l15) * K + half * 16;
  const bf16_t* bcol0 = BT + (size_t)(ct * 64 + l15) * K + half * 16;
  v8f acc0 = {}, acc1 = {}, acc2 = {}, acc3 = {};
  for (int k0 = 0; k0 < K; k0 += 32) {
    v16bf a = load16(arow + k0, arow + k0 + 8);
    const bf16_t* bp = bcol0 + k0;
    v16bf b0 = load16(bp,                   bp + 8);
    v16bf b1 = load16(bp + 16 * (size_t)K,  bp + 16 * (size_t)K + 8);
    v16bf b2 = load16(bp + 32 * (size_t)K,  bp + 32 * (size_t)K + 8);
    v16bf b3 = load16(bp + 48 * (size_t)K,  bp + 48 * (size_t)K + 8);
    acc0 = __builtin_amdgcn_wmma_f32_16x16x32_bf16(false, a, false, b0, (short)0, acc0, false, false);
    acc1 = __builtin_amdgcn_wmma_f32_16x16x32_bf16(false, a, false, b1, (short)0, acc1, false, false);
    acc2 = __builtin_amdgcn_wmma_f32_16x16x32_bf16(false, a, false, b2, (short)0, acc2, false, false);
    acc3 = __builtin_amdgcn_wmma_f32_16x16x32_bf16(false, a, false, b3, (short)0, acc3, false, false);
  }
  int rbase = mt * 16 + half * 8;
  int cbase = ct * 64 + l15;
#pragma unroll
  for (int v = 0; v < 8; ++v) {
    size_t r = (size_t)(rbase + v) * Ncols;
    C[r + cbase +  0] = acc0[v];
    C[r + cbase + 16] = acc1[v];
    C[r + cbase + 32] = acc2[v];
    C[r + cbase + 48] = acc3[v];
  }
}

// ---------------------------------------------------------------------------
// GAT attention pieces
// ---------------------------------------------------------------------------
__global__ void attn_coef_k(const float* __restrict__ h, const float* __restrict__ a_s,
                            const float* __restrict__ a_d, float* __restrict__ es,
                            float* __restrict__ ed, int n, int H, int F) {
  long long i = (long long)blockIdx.x * blockDim.x + threadIdx.x;
  if (i >= (long long)n * H) return;
  int node = (int)(i / H), hd = (int)(i % H);
  const float* hp = h + (size_t)node * H * F + (size_t)hd * F;
  float s = 0.f, d = 0.f;
  for (int f = 0; f < F; ++f) { float v = hp[f]; s += v * a_s[hd * F + f]; d += v * a_d[hd * F + f]; }
  es[i] = s;
  ed[i] = d;
}

__device__ __forceinline__ unsigned fkey_enc(float v) {
  unsigned u = __float_as_uint(v);
  return (u & 0x80000000u) ? ~u : (u | 0x80000000u);
}
__device__ __forceinline__ float fkey_dec(unsigned k) {
  return (k & 0x80000000u) ? __uint_as_float(k & 0x7FFFFFFFu) : __uint_as_float(~k);
}

// edges [0,E) are real, [E, E+n) are self-loops
__global__ void edge_max_k(const int* __restrict__ src, const int* __restrict__ dst,
                           const float* __restrict__ es, const float* __restrict__ ed,
                           unsigned* __restrict__ mkey, int E, int n, int H) {
  long long i = (long long)blockIdx.x * blockDim.x + threadIdx.x;
  long long total = (long long)(E + n) * H;
  if (i >= total) return;
  int e = (int)(i / H), hd = (int)(i % H);
  int s = (e < E) ? src[e] : (e - E);
  int d = (e < E) ? dst[e] : (e - E);
  float v = es[(size_t)s * H + hd] + ed[(size_t)d * H + hd];
  v = (v > 0.f) ? v : 0.2f * v;                    // leaky_relu(0.2)
  atomicMax(&mkey[(size_t)d * H + hd], fkey_enc(v));
}

__global__ void edge_exp_k(const int* __restrict__ src, const int* __restrict__ dst,
                           const float* __restrict__ es, const float* __restrict__ ed,
                           const unsigned* __restrict__ mkey, float* __restrict__ exbuf,
                           float* __restrict__ denom, int E, int n, int H) {
  long long i = (long long)blockIdx.x * blockDim.x + threadIdx.x;
  long long total = (long long)(E + n) * H;
  if (i >= total) return;
  int e = (int)(i / H), hd = (int)(i % H);
  int s = (e < E) ? src[e] : (e - E);
  int d = (e < E) ? dst[e] : (e - E);
  float v = es[(size_t)s * H + hd] + ed[(size_t)d * H + hd];
  v = (v > 0.f) ? v : 0.2f * v;
  float m = fkey_dec(mkey[(size_t)d * H + hd]);
  float ex = __expf(v - m);
  exbuf[i] = ex;
  atomicAdd(&denom[(size_t)d * H + hd], ex);
}

// grid = E+n blocks, block = H*F threads
__global__ void edge_aggr_k(const int* __restrict__ src, const int* __restrict__ dst,
                            const float* __restrict__ hfeat, const float* __restrict__ exbuf,
                            const float* __restrict__ denom, float* __restrict__ out,
                            int E, int H, int F) {
  int e = blockIdx.x;
  int f = threadIdx.x;            // 0..H*F-1
  int hd = f / F;
  int s = (e < E) ? src[e] : (e - E);
  int d = (e < E) ? dst[e] : (e - E);
  float alpha = exbuf[(size_t)e * H + hd] / (denom[(size_t)d * H + hd] + 1e-16f);
  atomicAdd(&out[(size_t)d * H * F + f], hfeat[(size_t)s * H * F + f] * alpha);
}

__global__ void bias_relu_bf16_k(const float* __restrict__ in, const float* __restrict__ bias,
                                 bf16_t* __restrict__ out, long long n, int C) {
  long long i = (long long)blockIdx.x * blockDim.x + threadIdx.x;
  if (i >= n) return;
  int c = (int)(i % C);
  float v = in[i] + bias[c];
  out[i] = f2bf(v > 0.f ? v : 0.f);
}

__global__ void bias_relu_f32_k(float* __restrict__ io, const float* __restrict__ bias,
                                long long n, int C) {
  long long i = (long long)blockIdx.x * blockDim.x + threadIdx.x;
  if (i >= n) return;
  int c = (int)(i % C);
  float v = io[i] + bias[c];
  io[i] = v > 0.f ? v : 0.f;
}

// ---------------------------------------------------------------------------
// Pooling + batchnorm
// ---------------------------------------------------------------------------
__global__ void pool_sum_k(const float* __restrict__ x, const int* __restrict__ batch,
                           float* __restrict__ psum, float* __restrict__ cnt, int n, int C) {
  long long i = (long long)blockIdx.x * blockDim.x + threadIdx.x;
  if (i >= (long long)n * C) return;
  int node = (int)(i / C), c = (int)(i % C);
  int g = batch[node];
  atomicAdd(&psum[(size_t)g * C + c], x[i]);
  if (c == 0) atomicAdd(&cnt[g], 1.f);
}

__global__ void pool_mean_k(float* __restrict__ psum, const float* __restrict__ cnt,
                            int G_, int C) {
  long long i = (long long)blockIdx.x * blockDim.x + threadIdx.x;
  if (i >= (long long)G_ * C) return;
  int g = (int)(i / C);
  psum[i] /= fmaxf(cnt[g], 1.f);
}

__global__ void bn_stats_k(const float* __restrict__ p, float* __restrict__ mu,
                           float* __restrict__ var, int G_, int C) {
  __shared__ float ss[256], sq[256];
  int f = blockIdx.x;
  float s = 0.f, q = 0.f;
  for (int g = threadIdx.x; g < G_; g += 256) {
    float v = p[(size_t)g * C + f];
    s += v; q += v * v;
  }
  ss[threadIdx.x] = s; sq[threadIdx.x] = q;
  __syncthreads();
  for (int st = 128; st > 0; st >>= 1) {
    if ((int)threadIdx.x < st) { ss[threadIdx.x] += ss[threadIdx.x + st]; sq[threadIdx.x] += sq[threadIdx.x + st]; }
    __syncthreads();
  }
  if (threadIdx.x == 0) {
    float m = ss[0] / (float)G_;
    mu[f] = m;
    var[f] = sq[0] / (float)G_ - m * m;
  }
}

__global__ void bn_apply_k(float* __restrict__ p, const float* __restrict__ mu,
                           const float* __restrict__ var, const float* __restrict__ gw,
                           const float* __restrict__ gb, int G_, int C) {
  long long i = (long long)blockIdx.x * blockDim.x + threadIdx.x;
  if (i >= (long long)G_ * C) return;
  int c = (int)(i % C);
  p[i] = gw[c] * (p[i] - mu[c]) * rsqrtf(var[c] + 1e-5f) + gb[c];
}

// ---------------------------------------------------------------------------
// LSTM / sequence / MLP
// ---------------------------------------------------------------------------
__global__ void build_seq_k(const float* __restrict__ eA, const float* __restrict__ eB,
                            float* __restrict__ seq, int G_, int C) {
  long long i = (long long)blockIdx.x * blockDim.x + threadIdx.x;
  if (i >= (long long)G_ * C) return;
  int g = (int)(i / C), c = (int)(i % C);
  seq[(size_t)g * 2 * C + c]     = eA[i];
  seq[(size_t)g * 2 * C + C + c] = eB[i];
}

// grid = 2*G blocks (graph, dir), block = 64 threads (hidden units)
__global__ void __launch_bounds__(64) lstm_layer_k(
    const float* __restrict__ seq, const float* __restrict__ Wih,
    const float* __restrict__ Whh, const float* __restrict__ bih,
    const float* __restrict__ bhh, float* __restrict__ out,
    float* __restrict__ hn, int G_) {
  int g = blockIdx.x >> 1;
  int dir = blockIdx.x & 1;
  int u = threadIdx.x;
  const float* wih = Wih + (size_t)dir * 256 * 128;
  const float* whh = Whh + (size_t)dir * 256 * 64;
  const float* bi  = bih + dir * 256;
  const float* bh  = bhh + dir * 256;
  __shared__ float xs[2][128];
  __shared__ float hs[64];
  for (int i = u; i < 256; i += 64) xs[i >> 7][i & 127] = seq[(size_t)g * 256 + i];
  hs[u] = 0.f;
  float c = 0.f, h = 0.f;
  __syncthreads();
  for (int step = 0; step < 2; ++step) {
    int t = dir ? (1 - step) : step;
    float gi = bi[u]       + bh[u];
    float gf = bi[64 + u]  + bh[64 + u];
    float gg = bi[128 + u] + bh[128 + u];
    float go = bi[192 + u] + bh[192 + u];
    for (int d = 0; d < 128; ++d) {
      float x = xs[t][d];
      gi += wih[(size_t)u * 128 + d] * x;
      gf += wih[(size_t)(64 + u) * 128 + d] * x;
      gg += wih[(size_t)(128 + u) * 128 + d] * x;
      go += wih[(size_t)(192 + u) * 128 + d] * x;
    }
    for (int d = 0; d < 64; ++d) {
      float hv = hs[d];
      gi += whh[(size_t)u * 64 + d] * hv;
      gf += whh[(size_t)(64 + u) * 64 + d] * hv;
      gg += whh[(size_t)(128 + u) * 64 + d] * hv;
      go += whh[(size_t)(192 + u) * 64 + d] * hv;
    }
    c = sigmoidf_(gf) * c + sigmoidf_(gi) * tanhf(gg);
    h = sigmoidf_(go) * tanhf(c);
    __syncthreads();
    hs[u] = h;
    out[((size_t)g * 2 + t) * 128 + dir * 64 + u] = h;
    __syncthreads();
  }
  hn[(size_t)g * 128 + dir * 64 + u] = h;
}

__global__ void dense_k(const float* __restrict__ in, const float* __restrict__ W,
                        const float* __restrict__ b, float* __restrict__ out,
                        int rows, int Din, int Dout, int relu) {
  long long i = (long long)blockIdx.x * blockDim.x + threadIdx.x;
  if (i >= (long long)rows * Dout) return;
  int r = (int)(i / Dout), j = (int)(i % Dout);
  float s = b[j];
  const float* ip = in + (size_t)r * Din;
  for (int d = 0; d < Din; ++d) s += ip[d] * W[(size_t)d * Dout + j];
  if (relu) s = fmaxf(s, 0.f);
  out[i] = s;
}

// ---------------------------------------------------------------------------
// Host side
// ---------------------------------------------------------------------------
static void fill_u32(void* p, unsigned v, long long n, hipStream_t s) {
  long long b = (n + 255) / 256;
  int blocks = (int)(b < 4096 ? (b < 1 ? 1 : b) : 4096);
  fill_u32_k<<<blocks, 256, 0, s>>>((unsigned*)p, v, n);
}

static const unsigned NEG_INF_KEY = ~0xFF800000u; // fkey_enc(-inf)

struct Scratch {
  bf16_t *W1T, *W2T, *xbf;
  float *h1, *out1;
  float *es, *ed, *denom;
  unsigned *mkey;
  float *ex;
};

static void encode_side(const float* x, const int* ei, const int* batch,
                        const float* as1, const float* ad1, const float* b1,
                        const float* as2, const float* ad2, const float* b2,
                        const float* bng, const float* bnb,
                        float* psum, float* cnt, float* mu, float* var,
                        const Scratch& S, int N, int E, int G_, hipStream_t st) {
  const int* src = ei;
  const int* dst = ei + E;
  const int EL = E + N;

  // ---- Layer 1: linear (WMMA), 58->256 (Kpad 64), 4 heads x 64 ----
  cvt_pad_bf16_k<<<cdiv_i((long long)N * 64, 256), 256, 0, st>>>(x, S.xbf, N, NODE_DIM_, 64);
  {
    long long waves = (long long)(N / 16) * (256 / 64);
    wmma_gemm_bf16_k<<<cdiv_i(waves * 32, 256), 256, 0, st>>>(S.xbf, S.W1T, S.h1, N, 64, 256);
  }
  attn_coef_k<<<cdiv_i((long long)N * 4, 256), 256, 0, st>>>(S.h1, as1, ad1, S.es, S.ed, N, 4, 64);
  fill_u32(S.mkey, NEG_INF_KEY, (long long)N * 4, st);
  fill_u32(S.denom, 0u, (long long)N * 4, st);
  fill_u32(S.out1, 0u, (long long)N * 256, st);
  edge_max_k<<<cdiv_i((long long)EL * 4, 256), 256, 0, st>>>(src, dst, S.es, S.ed, S.mkey, E, N, 4);
  edge_exp_k<<<cdiv_i((long long)EL * 4, 256), 256, 0, st>>>(src, dst, S.es, S.ed, S.mkey, S.ex, S.denom, E, N, 4);
  edge_aggr_k<<<EL, 256, 0, st>>>(src, dst, S.h1, S.ex, S.denom, S.out1, E, 4, 64);

  // relu(out1 + b1) -> bf16 activations (reuse h1 region)
  bf16_t* h1bf = (bf16_t*)S.h1;
  bias_relu_bf16_k<<<cdiv_i((long long)N * 256, 256), 256, 0, st>>>(S.out1, b1, h1bf, (long long)N * 256, 256);

  // ---- Layer 2: linear (WMMA), 256->128, 1 head ----
  float* h2 = S.out1;                                      // reuse (out1 dead)
  {
    long long waves = (long long)(N / 16) * (128 / 64);
    wmma_gemm_bf16_k<<<cdiv_i(waves * 32, 256), 256, 0, st>>>(h1bf, S.W2T, h2, N, 256, 128);
  }
  attn_coef_k<<<cdiv_i((long long)N, 256), 256, 0, st>>>(h2, as2, ad2, S.es, S.ed, N, 1, 128);
  float* out2 = (float*)((char*)S.h1 + (size_t)N * 256 * 2); // second half of h1 region
  fill_u32(S.mkey, NEG_INF_KEY, N, st);
  fill_u32(S.denom, 0u, N, st);
  fill_u32(out2, 0u, (long long)N * 128, st);
  edge_max_k<<<cdiv_i((long long)EL, 256), 256, 0, st>>>(src, dst, S.es, S.ed, S.mkey, E, N, 1);
  edge_exp_k<<<cdiv_i((long long)EL, 256), 256, 0, st>>>(src, dst, S.es, S.ed, S.mkey, S.ex, S.denom, E, N, 1);
  edge_aggr_k<<<EL, 128, 0, st>>>(src, dst, h2, S.ex, S.denom, out2, E, 1, 128);
  bias_relu_f32_k<<<cdiv_i((long long)N * 128, 256), 256, 0, st>>>(out2, b2, (long long)N * 128, 128);

  // ---- global mean pool + batchnorm (per side) ----
  fill_u32(psum, 0u, (long long)G_ * 128, st);
  fill_u32(cnt, 0u, G_, st);
  pool_sum_k<<<cdiv_i((long long)N * 128, 256), 256, 0, st>>>(out2, batch, psum, cnt, N, 128);
  pool_mean_k<<<cdiv_i((long long)G_ * 128, 256), 256, 0, st>>>(psum, cnt, G_, 128);
  bn_stats_k<<<128, 256, 0, st>>>(psum, mu, var, G_, 128);
  bn_apply_k<<<cdiv_i((long long)G_ * 128, 256), 256, 0, st>>>(psum, mu, var, bng, bnb, G_, 128);
}

extern "C" void kernel_launch(void* const* d_in, const int* in_sizes, int n_in,
                              void* d_out, int out_size, void* d_ws, size_t ws_size,
                              hipStream_t stream) {
  (void)n_in; (void)ws_size;
  const float* xA     = (const float*)d_in[0];
  const int*   eiA    = (const int*)d_in[1];
  const int*   batchA = (const int*)d_in[2];
  const float* xB     = (const float*)d_in[3];
  const int*   eiB    = (const int*)d_in[4];
  const int*   batchB = (const int*)d_in[5];
  const float* W1  = (const float*)d_in[6];
  const float* as1 = (const float*)d_in[7];
  const float* ad1 = (const float*)d_in[8];
  const float* b1  = (const float*)d_in[9];
  const float* W2  = (const float*)d_in[10];
  const float* as2 = (const float*)d_in[11];
  const float* ad2 = (const float*)d_in[12];
  const float* b2  = (const float*)d_in[13];
  const float* bng = (const float*)d_in[14];
  const float* bnb = (const float*)d_in[15];
  const float* Wih = (const float*)d_in[16];
  const float* Whh = (const float*)d_in[17];
  const float* bih = (const float*)d_in[18];
  const float* bhh = (const float*)d_in[19];
  const float* h1W = (const float*)d_in[20];
  const float* h1b = (const float*)d_in[21];
  const float* h2W = (const float*)d_in[22];
  const float* h2b = (const float*)d_in[23];
  const float* h3W = (const float*)d_in[24];
  const float* h3b = (const float*)d_in[25];

  const int N  = in_sizes[0] / NODE_DIM_;
  const int E  = in_sizes[1] / 2;
  const int G_ = out_size / 2;
  const int EL = E + N;

  // ---- workspace layout ----
  char* base = (char*)d_ws;
  size_t off = 0;
  auto alloc = [&](size_t bytes) -> void* {
    off = (off + 255) & ~(size_t)255;
    void* p = base + off;
    off += bytes;
    return p;
  };
  Scratch S;
  S.W1T   = (bf16_t*)alloc((size_t)256 * 64 * 2);
  S.W2T   = (bf16_t*)alloc((size_t)128 * 256 * 2);
  S.xbf   = (bf16_t*)alloc((size_t)N * 64 * 2);
  S.h1    = (float*)alloc((size_t)N * 256 * 4);     // reused: h1bf (1st half), out2 (2nd half)
  S.out1  = (float*)alloc((size_t)N * 256 * 4);     // reused: h2
  S.es    = (float*)alloc((size_t)N * 4 * 4);
  S.ed    = (float*)alloc((size_t)N * 4 * 4);
  S.denom = (float*)alloc((size_t)N * 4 * 4);
  S.mkey  = (unsigned*)alloc((size_t)N * 4 * 4);
  S.ex    = (float*)alloc((size_t)EL * 4 * 4);
  float* psumA = (float*)alloc((size_t)G_ * 128 * 4);
  float* psumB = (float*)alloc((size_t)G_ * 128 * 4);
  float* cnt   = (float*)alloc((size_t)G_ * 4);
  float* mu    = (float*)alloc(128 * 4);
  float* var   = (float*)alloc(128 * 4);
  float* seq1  = (float*)alloc((size_t)G_ * 2 * 128 * 4);
  float* seq2  = (float*)alloc((size_t)G_ * 2 * 128 * 4);
  float* hn    = (float*)alloc((size_t)G_ * 128 * 4);
  float* m1    = (float*)alloc((size_t)G_ * 128 * 4);
  float* m2    = (float*)alloc((size_t)G_ * 64 * 4);

  // ---- weight prep (bf16, transposed for WMMA B operand) ----
  cvt_wt_k<<<cdiv_i((long long)256 * 64, 256), 256, 0, stream>>>(W1, S.W1T, NODE_DIM_, 256, 64);
  cvt_wt_k<<<cdiv_i((long long)128 * 256, 256), 256, 0, stream>>>(W2, S.W2T, 256, 128, 256);

  // ---- Siamese GAT encoders ----
  encode_side(xA, eiA, batchA, as1, ad1, b1, as2, ad2, b2, bng, bnb,
              psumA, cnt, mu, var, S, N, E, G_, stream);
  encode_side(xB, eiB, batchB, as1, ad1, b1, as2, ad2, b2, bng, bnb,
              psumB, cnt, mu, var, S, N, E, G_, stream);

  // ---- BiLSTM (2 layers, T=2) ----
  build_seq_k<<<cdiv_i((long long)G_ * 128, 256), 256, 0, stream>>>(psumA, psumB, seq1, G_, 128);
  lstm_layer_k<<<2 * G_, 64, 0, stream>>>(seq1, Wih, Whh, bih, bhh, seq2, hn, G_);
  lstm_layer_k<<<2 * G_, 64, 0, stream>>>(seq2,
                                          Wih + (size_t)2 * 256 * 128,
                                          Whh + (size_t)2 * 256 * 64,
                                          bih + 2 * 256, bhh + 2 * 256,
                                          seq1, hn, G_);

  // ---- MLP head ----
  dense_k<<<cdiv_i((long long)G_ * 128, 256), 256, 0, stream>>>(hn, h1W, h1b, m1, G_, 128, 128, 1);
  dense_k<<<cdiv_i((long long)G_ * 64, 256), 256, 0, stream>>>(m1, h2W, h2b, m2, G_, 128, 64, 1);
  dense_k<<<cdiv_i((long long)G_ * 2, 256), 256, 0, stream>>>(m2, h3W, h3b, (float*)d_out, G_, 64, 2, 0);
}
